// RFP_38568806318080
// MI455X (gfx1250) — matrix operand order repaired and checked by reading, not compiled, not run on hardware
//
#include <hip/hip_runtime.h>

// ---------------- problem constants ----------------
enum : int { BB = 256, SS = 64, DD = 256, HH = 1024, OO = 512, G3 = 3 * HH };

// ---------------- types ----------------
typedef __attribute__((ext_vector_type(16))) __bf16 v16bf;
typedef __attribute__((ext_vector_type(8)))  float  v8f;
typedef unsigned int u32x4 __attribute__((ext_vector_type(4)));
typedef unsigned int u32x8 __attribute__((ext_vector_type(8)));

union Frag { u32x4 q[2]; v16bf v; };

__device__ __forceinline__ unsigned short f2bf_rne(float f) {
  unsigned int x = __float_as_uint(f);
  unsigned int r = x + 0x7fffu + ((x >> 16) & 1u);   // round-to-nearest-even
  return (unsigned short)(r >> 16);
}

// ---------------- conversion / utility kernels ----------------
__global__ void cvt_f32_bf16(const float* __restrict__ s, unsigned short* __restrict__ d, int n) {
  int i = blockIdx.x * blockDim.x + threadIdx.x;
  if (i < n) d[i] = f2bf_rne(s[i]);
}

// lin_w: [S][H][O] fp32  ->  [S][O][H] bf16 (same layout as the other weights: W[N][K])
__global__ void linw_tr_bf16(const float* __restrict__ s, unsigned short* __restrict__ d, int n) {
  int i = blockIdx.x * blockDim.x + threadIdx.x;
  if (i >= n) return;
  int o  = i % OO;
  int r  = i / OO;
  int h  = r % HH;
  int t  = r / HH;
  d[((size_t)t * OO + o) * HH + h] = f2bf_rne(s[i]);
}

__global__ void zero_state(float* __restrict__ h, unsigned short* __restrict__ hb, int n) {
  int i = blockIdx.x * blockDim.x + threadIdx.x;
  if (i < n) { h[i] = 0.0f; hb[i] = 0; }
}

__global__ void copy_f32(const float* __restrict__ s, float* __restrict__ d, int n) {
  int i = blockIdx.x * blockDim.x + threadIdx.x;
  if (i < n) d[i] = s[i];
}

// ---------------- async data movers ----------------
// TDM: load one 128-row x 32-elem bf16 weight tile (row stride K elems) into LDS,
// with HW padding of 4 DWORDs after every 16 DWORDs -> 80B LDS row stride.
// D# layout per CDNA5 ISA 8.3/8.4. Issued by one wave; tracked on TENSORcnt.
__device__ __forceinline__ void tdm_load_b_tile(const unsigned short* gsrc,
                                                unsigned int lds_off, int K) {
  unsigned long long ga = (unsigned long long)(size_t)gsrc;
  u32x4 g0;
  g0[0] = 1u;                                            // count=1, user descriptor
  g0[1] = lds_off;                                       // lds_addr (bytes)
  g0[2] = (unsigned int)ga;                              // global_addr[31:0]
  g0[3] = ((unsigned int)(ga >> 32) & 0x01ffffffu)       // global_addr[56:32]
          | 0x80000000u;                                 // type=2 ("image")
  u32x8 g1;
  g1[0] = (1u << 16)      // data_size = 2 bytes
        | (1u << 20)      // pad_enable
        | (3u << 22)      // pad_interval: 16 DWORDs (64B line)
        | (3u << 25);     // pad_amount:   4 DWORDs (16B)
  g1[1] = 32u << 16;      // tensor_dim0 = 32 (bits 79:48, low half)
  g1[2] = 128u << 16;     // tensor_dim1 = 128 (bits 111:80, low half)
  g1[3] = 32u << 16;      // tile_dim0 = 32 (bits 127:112)
  g1[4] = 128u;           // tile_dim1 = 128 (bits 143:128); tile_dim2 = 0
  g1[5] = (unsigned int)K;// tensor_dim0_stride[31:0] = K elems
  g1[6] = 0u;             // stride0 hi, stride1 lo
  g1[7] = 0u;             // stride1 hi
  asm volatile("tensor_load_to_lds %0, %1" :: "s"(g0), "s"(g1) : "memory");
}

// Per-lane async 16B chunk: global -> LDS, tracked on ASYNCcnt (no VGPR data path).
__device__ __forceinline__ void async_load_a_chunk(const unsigned short* gsrc,
                                                   unsigned int lds_off) {
  unsigned long long ga = (unsigned long long)(size_t)gsrc;
  asm volatile("global_load_async_to_lds_b128 %0, %1, off"
               :: "v"(lds_off), "v"(ga) : "memory");
}

// ---------------- WMMA bf16 GEMM:  C[M,N] = A[M,K] * W[N,K]^T (+bias) ----------------
// Block tile 64(M) x 128(N), K-tile 32, double-buffered LDS fed by async/TDM DMA.
// 256 threads = 8 waves; each wave owns a 32x32 sub-tile (2x2 wmma accumulators).
#define TM 64
#define TN 128
#define TK 32

__global__ __launch_bounds__(256)
void gemm_bf16_wmma(const unsigned short* __restrict__ A, int K, int lda,
                    const unsigned short* __restrict__ W,
                    const float* __restrict__ bias,
                    float* __restrict__ C, long long ldc) {
  __shared__ unsigned short As[2][TM][TK + 8];   // 80B row stride (16B aligned)
  __shared__ unsigned short Bs[2][TN][TK + 8];

  const int tid   = threadIdx.x;
  const int Mbase = blockIdx.y * TM;
  const int Nbase = blockIdx.x * TN;
  const int lane  = tid & 31;
  const int wid   = tid >> 5;
  const int wm    = wid & 1;        // M sub-offset 32*wm
  const int wn    = wid >> 1;       // N sub-offset 32*wn
  const int half  = lane >> 4;
  const int l16   = lane & 15;

  const v8f vzero = {0.f, 0.f, 0.f, 0.f, 0.f, 0.f, 0.f, 0.f};
  v8f acc[2][2];
  acc[0][0] = vzero; acc[0][1] = vzero; acc[1][0] = vzero; acc[1][1] = vzero;

  // LDS byte offsets (workgroup-relative: low 32 bits of the flat LDS address)
  const unsigned int asBase   = (unsigned int)(size_t)&As[0][0][0];
  const unsigned int bsBase   = (unsigned int)(size_t)&Bs[0][0][0];
  const unsigned int asStride = (unsigned int)sizeof(As[0]);
  const unsigned int bsStride = (unsigned int)sizeof(Bs[0]);

  // per-thread A chunk: 64 rows x 4 chunks of 8 bf16
  const int arow = tid >> 2;
  const int acol = (tid & 3) * 8;
  const unsigned int aOff = (unsigned int)(arow * (TK + 8) + acol) * 2u;
  const unsigned short* Arow = A + (size_t)(Mbase + arow) * lda + acol;

  const int nk = K / TK;

  // ---- prologue: DMA tile 0 into buffer 0 ----
  async_load_a_chunk(Arow, asBase + aOff);
  if (wid == 0) tdm_load_b_tile(W + (size_t)Nbase * K, bsBase, K);
  asm volatile("s_wait_asynccnt 0x0" ::: "memory");
  if (wid == 0) __builtin_amdgcn_s_wait_tensorcnt(0);
  __syncthreads();

  for (int kt = 0; kt < nk; ++kt) {
    const int buf = kt & 1;
    // ---- issue DMA for tile kt+1 into the other buffer (overlaps compute) ----
    if (kt + 1 < nk) {
      const int k0n = (kt + 1) * TK;
      async_load_a_chunk(Arow + k0n, asBase + (buf ^ 1) * asStride + aOff);
      if (wid == 0)
        tdm_load_b_tile(W + (size_t)Nbase * K + k0n, bsBase + (buf ^ 1) * bsStride, K);
    }

    // ---- fragments per ISA 7.12.2 (A and B: K = 16*(i/8)+8*half+i%8) ----
    Frag a[2], b[2];
#pragma unroll
    for (int mi = 0; mi < 2; ++mi) {
      int r = wm * 32 + mi * 16 + l16;
      a[mi].q[0] = *(const u32x4*)&As[buf][r][8 * half];
      a[mi].q[1] = *(const u32x4*)&As[buf][r][16 + 8 * half];
    }
#pragma unroll
    for (int ni = 0; ni < 2; ++ni) {
      int r = wn * 32 + ni * 16 + l16;
      b[ni].q[0] = *(const u32x4*)&Bs[buf][r][8 * half];
      b[ni].q[1] = *(const u32x4*)&Bs[buf][r][16 + 8 * half];
    }

#pragma unroll
    for (int mi = 0; mi < 2; ++mi)
#pragma unroll
      for (int ni = 0; ni < 2; ++ni)
        acc[mi][ni] = __builtin_amdgcn_wmma_f32_16x16x32_bf16(
            false, a[mi].v, false, b[ni].v, (short)0, acc[mi][ni], false, false);

    // ---- next tile's DMA must have landed before the next iteration reads it ----
    asm volatile("s_wait_asynccnt 0x0" ::: "memory");
    if (wid == 0) __builtin_amdgcn_s_wait_tensorcnt(0);
    __syncthreads();
  }

  // ---- epilogue: C/D layout — VGPR e: row = e + 8*half, col = l16 ----
#pragma unroll
  for (int mi = 0; mi < 2; ++mi)
#pragma unroll
    for (int ni = 0; ni < 2; ++ni) {
      const int col  = Nbase + wn * 32 + ni * 16 + l16;
      const float bv = bias ? bias[col] : 0.0f;
#pragma unroll
      for (int e = 0; e < 8; ++e) {
        const int row = Mbase + wm * 32 + mi * 16 + 8 * half + e;
        C[(long long)row * ldc + col] = acc[mi][ni][e] + bv;
      }
    }
}

// ---------------- GRU elementwise: h' = (1-z)*n + z*h  (in-place fp32 + bf16 copy) ----------------
__global__ void gru_ew(const float* __restrict__ gi, const float* __restrict__ gh,
                       const float* __restrict__ bih, const float* __restrict__ bhh,
                       float* __restrict__ h, unsigned short* __restrict__ hb) {
  int i = blockIdx.x * blockDim.x + threadIdx.x;   // 0 .. B*H-1
  int b = i >> 10;
  int j = i & (HH - 1);
  size_t base = (size_t)b * G3;
  float ir = gi[base + j]          + bih[j];
  float iz = gi[base + HH + j]     + bih[HH + j];
  float in = gi[base + 2 * HH + j] + bih[2 * HH + j];
  float hr = gh[base + j]          + bhh[j];
  float hz = gh[base + HH + j]     + bhh[HH + j];
  float hn = gh[base + 2 * HH + j] + bhh[2 * HH + j];
  float r = 1.0f / (1.0f + expf(-(ir + hr)));
  float z = 1.0f / (1.0f + expf(-(iz + hz)));
  float n = tanhf(in + r * hn);
  float hv = (1.0f - z) * n + z * h[i];
  h[i]  = hv;
  hb[i] = f2bf_rne(hv);
}

// ---------------- host driver ----------------
extern "C" void kernel_launch(void* const* d_in, const int* in_sizes, int n_in,
                              void* d_out, int out_size, void* d_ws, size_t ws_size,
                              hipStream_t stream) {
  const float* x       = (const float*)d_in[0];
  const float* wih[3]  = {(const float*)d_in[1], (const float*)d_in[5], (const float*)d_in[9]};
  const float* whh[3]  = {(const float*)d_in[2], (const float*)d_in[6], (const float*)d_in[10]};
  const float* bih[3]  = {(const float*)d_in[3], (const float*)d_in[7], (const float*)d_in[11]};
  const float* bhh[3]  = {(const float*)d_in[4], (const float*)d_in[8], (const float*)d_in[12]};
  const float* linw    = (const float*)d_in[13];
  const float* linb    = (const float*)d_in[14];
  float* out = (float*)d_out;

  // ---- workspace carve (~115 MB, 256B aligned slices) ----
  char* p = (char*)d_ws;
  auto carve = [&](size_t bytes) -> char* {
    char* r = p; p += (bytes + 255) & ~(size_t)255; return r;
  };
  unsigned short* xb     = (unsigned short*)carve((size_t)BB * SS * DD * 2);
  unsigned short* wihb[3], *whhb[3];
  wihb[0] = (unsigned short*)carve((size_t)G3 * DD * 2);
  whhb[0] = (unsigned short*)carve((size_t)G3 * HH * 2);
  for (int c = 1; c < 3; ++c) {
    wihb[c] = (unsigned short*)carve((size_t)G3 * HH * 2);
    whhb[c] = (unsigned short*)carve((size_t)G3 * HH * 2);
  }
  unsigned short* linwt = (unsigned short*)carve((size_t)SS * OO * HH * 2);
  float*          hS    = (float*)carve((size_t)3 * BB * HH * 4);          // h0|h1|h2 fp32
  unsigned short* hBf   = (unsigned short*)carve((size_t)3 * BB * HH * 2); // bf16 mirrors
  float*          gi    = (float*)carve((size_t)BB * G3 * 4);
  float*          gh    = (float*)carve((size_t)BB * G3 * 4);
  (void)ws_size; (void)n_in; (void)in_sizes; (void)out_size;

  const int T = 256;
  auto nb = [](int n) { return (n + 255) / 256; };

  // ---- one-time (per launch) conversions to bf16 ----
  cvt_f32_bf16<<<nb(BB * SS * DD), T, 0, stream>>>(x, xb, BB * SS * DD);
  cvt_f32_bf16<<<nb(G3 * DD), T, 0, stream>>>(wih[0], wihb[0], G3 * DD);
  cvt_f32_bf16<<<nb(G3 * HH), T, 0, stream>>>(whh[0], whhb[0], G3 * HH);
  for (int c = 1; c < 3; ++c) {
    cvt_f32_bf16<<<nb(G3 * HH), T, 0, stream>>>(wih[c], wihb[c], G3 * HH);
    cvt_f32_bf16<<<nb(G3 * HH), T, 0, stream>>>(whh[c], whhb[c], G3 * HH);
  }
  linw_tr_bf16<<<nb(SS * HH * OO), T, 0, stream>>>(linw, linwt, SS * HH * OO);
  zero_state<<<nb(3 * BB * HH), T, 0, stream>>>(hS, hBf, 3 * BB * HH);

  const dim3 gridGate(G3 / TN, BB / TM);   // (24, 4)
  const dim3 gridLin (OO / TN, BB / TM);   // (4, 4)

  // ---- 64 sequential steps (kernel chain = global barrier between stages) ----
  for (int t = 0; t < SS; ++t) {
    for (int c = 0; c < 3; ++c) {
      const unsigned short* Ain = (c == 0) ? (xb + (size_t)t * DD)
                                           : (hBf + (size_t)(c - 1) * BB * HH);
      const int Kin  = (c == 0) ? DD : HH;
      const int ldin = (c == 0) ? SS * DD : HH;
      gemm_bf16_wmma<<<gridGate, T, 0, stream>>>(Ain, Kin, ldin, wihb[c],
                                                 nullptr, gi, (long long)G3);
      gemm_bf16_wmma<<<gridGate, T, 0, stream>>>(hBf + (size_t)c * BB * HH, HH, HH,
                                                 whhb[c], nullptr, gh, (long long)G3);
      gru_ew<<<nb(BB * HH), T, 0, stream>>>(gi, gh, bih[c], bhh[c],
                                            hS + (size_t)c * BB * HH,
                                            hBf + (size_t)c * BB * HH);
    }
    // out[:, t, :] = h2 @ lin_w[t] + lin_b[t]   (lin_w pre-transposed to [O][H])
    gemm_bf16_wmma<<<gridLin, T, 0, stream>>>(hBf + (size_t)2 * BB * HH, HH, HH,
                                              linwt + (size_t)t * OO * HH,
                                              linb + (size_t)t * OO,
                                              out + (size_t)t * OO,
                                              (long long)SS * OO);
  }

  // ---- final hidden states appended after outs ----
  copy_f32<<<nb(3 * BB * HH), T, 0, stream>>>(hS, out + (size_t)BB * SS * OO, 3 * BB * HH);
}
